// GCNClassifier_40149354283623
// MI455X (gfx1250) — matrix operand order repaired and checked by reading, not compiled, not run on hardware
//
#include <hip/hip_runtime.h>
#include <math.h>

// ---------- types for WMMA ----------
typedef __attribute__((ext_vector_type(16))) __bf16 v16bf;
typedef __attribute__((ext_vector_type(8)))  float  v8f;

union Frag32B { uint4 q[2]; v16bf v; };

__device__ __forceinline__ unsigned short f2bf(float f) {
  unsigned int u = __float_as_uint(f);
  u += 0x7FFFu + ((u >> 16) & 1u);       // round-to-nearest-even
  return (unsigned short)(u >> 16);
}
__device__ __forceinline__ float bf2f(unsigned short h) {
  return __uint_as_float(((unsigned int)h) << 16);
}
__device__ __forceinline__ float leaky(float v) {
  return v >= 0.0f ? v : 0.01f * v;
}

// ---------- small utility kernels ----------
__global__ void k_zero_f32(float* __restrict__ p, long long n) {
  long long i = (long long)blockIdx.x * blockDim.x + threadIdx.x;
  if (i < n) p[i] = 0.0f;
}

__global__ void k_degree(const int* __restrict__ ei, float* __restrict__ deg, long long E) {
  long long i = (long long)blockIdx.x * blockDim.x + threadIdx.x;
  if (i < E) atomicAdd(&deg[ei[E + i]], 1.0f);
}

__global__ void k_dis(const float* __restrict__ deg, float* __restrict__ dis, int N) {
  int i = blockIdx.x * blockDim.x + threadIdx.x;
  if (i < N) dis[i] = rsqrtf(deg[i] + 1.0f);
}

__global__ void k_f32_to_bf16(const float* __restrict__ s, unsigned short* __restrict__ d,
                              long long n) {
  long long i = (long long)blockIdx.x * blockDim.x + threadIdx.x;
  if (i < n) d[i] = f2bf(s[i]);
}

// W[k][n] fp32 -> Wt[n][k] bf16 (column-major B for contiguous K-fragment loads)
__global__ void k_wtrans(const float* __restrict__ W, unsigned short* __restrict__ Wt,
                         int K, int Nout) {
  int id = blockIdx.x * blockDim.x + threadIdx.x;
  if (id < K * Nout) {
    int k = id / Nout;
    int n = id - k * Nout;
    Wt[(size_t)n * K + k] = f2bf(W[id]);
  }
}

// ---------- bf16 WMMA GEMM: C[M][Nout] = A[M][K] * B  (B given as Bt[Nout][K]) ----------
// block = 256 threads = 8 waves; block tile 64(M) x 64(N); each wave: 16(M) x 32(N).
__global__ __launch_bounds__(256) void k_gemm_bf16(
    const unsigned short* __restrict__ A,
    const unsigned short* __restrict__ Bt,
    float* __restrict__ C, int M, int K, int Nout)
{
  const int lane = threadIdx.x & 31;
  const int wave = threadIdx.x >> 5;
  const int rowBase = blockIdx.x * 64 + (wave & 3) * 16;
  const int colBase = blockIdx.y * 64 + (wave >> 2) * 32;

  // A fragment addressing (16-bit A 16x32 layout):
  // lane<16: row=lane, K = {0..7} and {16..23}; lane>=16: row=lane-16, K = {8..15} and {24..31}
  int row = rowBase + (lane & 15);
  if (row >= M) row = M - 1;               // clamp; stores are guarded below
  const unsigned short* arow = A + (size_t)row * K;
  const int kh = (lane & 16) ? 8 : 0;

  // B fragment addressing (32x16 layout): col = lane&15,
  // lanes 0-15 hold K = {0..7} and {16..23}; lanes 16-31 hold K = {8..15} and {24..31}
  const int kOff = (lane & 16) ? 8 : 0;
  const int c0 = colBase + (lane & 15);
  const unsigned short* bp0 = Bt + (size_t)c0 * K + kOff;
  const unsigned short* bp1 = Bt + (size_t)(c0 + 16) * K + kOff;

  v8f acc0 = {};
  v8f acc1 = {};

  for (int k0 = 0; k0 < K; k0 += 32) {
    Frag32B a, b0, b1;
    a.q[0]  = *(const uint4*)(arow + k0 + kh);
    a.q[1]  = *(const uint4*)(arow + k0 + 16 + kh);
    b0.q[0] = *(const uint4*)(bp0 + k0);
    b0.q[1] = *(const uint4*)(bp0 + k0 + 16);
    b1.q[0] = *(const uint4*)(bp1 + k0);
    b1.q[1] = *(const uint4*)(bp1 + k0 + 16);
    acc0 = __builtin_amdgcn_wmma_f32_16x16x32_bf16(false, a.v, false, b0.v,
                                                   (short)0, acc0, false, false);
    acc1 = __builtin_amdgcn_wmma_f32_16x16x32_bf16(false, a.v, false, b1.v,
                                                   (short)0, acc1, false, false);
  }

  // C/D layout: VGPR v -> M = v (lanes 0-15) or 8+v (lanes 16-31); N = lane&15
  const int rBase = rowBase + ((lane & 16) ? 8 : 0);
  const int cc = colBase + (lane & 15);
#pragma unroll
  for (int v = 0; v < 8; ++v) {
    int r = rBase + v;
    if (r < M) {
      C[(size_t)r * Nout + cc]      = acc0[v];
      C[(size_t)r * Nout + cc + 16] = acc1[v];
    }
  }
}

// agg = dis[i]^2 * xw + b[f]   (self-loop term + bias, also zero-initializes agg)
__global__ void k_init_agg(const float* __restrict__ xw, const float* __restrict__ dis,
                           const float* __restrict__ bias, float* __restrict__ agg,
                           long long total, int fshift) {
  long long id = (long long)blockIdx.x * blockDim.x + threadIdx.x;
  if (id >= total) return;
  int f = (int)(id & ((1LL << fshift) - 1));
  int i = (int)(id >> fshift);
  float d = dis[i];
  agg[id] = d * d * xw[id] + bias[f];
}

// one thread per (edge, 4-feature chunk): agg[dst] += en * xw[src]
__global__ __launch_bounds__(256) void k_scatter(
    const int* __restrict__ ei, const float* __restrict__ xw,
    const float* __restrict__ dis, float* __restrict__ agg,
    long long E, int cshift /* log2(F/4) */) {
  long long id = (long long)blockIdx.x * blockDim.x + threadIdx.x;
  if (id >= (E << cshift)) return;
  long long e = id >> cshift;
  int f4 = (int)(id & ((1LL << cshift) - 1));
  int F = 4 << cshift;
  int src = ei[e];
  int dst = ei[E + e];
  float en = dis[src] * dis[dst];
  const float4 v = *(const float4*)(xw + (size_t)src * F + (size_t)f4 * 4);
  float* o = agg + (size_t)dst * F + (size_t)f4 * 4;
  atomicAdd(o + 0, v.x * en);
  atomicAdd(o + 1, v.y * en);
  atomicAdd(o + 2, v.z * en);
  atomicAdd(o + 3, v.w * en);
}

__global__ void k_finalize(const float* __restrict__ agg, unsigned short* __restrict__ h,
                           long long n) {
  long long id = (long long)blockIdx.x * blockDim.x + threadIdx.x;
  if (id < n) h[id] = f2bf(leaky(agg[id]));
}

// mean-pool partial sums: block handles 256 rows, thread f accumulates, 1 atomic per (block,f)
__global__ __launch_bounds__(256) void k_pool(const unsigned short* __restrict__ h,
                                              float* __restrict__ pool, int N) {
  int f = threadIdx.x;          // Fout == 256
  int r0 = blockIdx.x * 256;
  int r1 = r0 + 256; if (r1 > N) r1 = N;
  float s = 0.0f;
  for (int r = r0; r < r1; ++r) s += bf2f(h[(size_t)r * 256 + f]);
  atomicAdd(&pool[f], s);
}

// MLP head: mean -> 256x64 -> leaky -> 64x1 -> sigmoid
__global__ __launch_bounds__(256) void k_head(const float* __restrict__ pool,
                                              const float* __restrict__ fcW1,
                                              const float* __restrict__ fcb1,
                                              const float* __restrict__ fcW2,
                                              const float* __restrict__ fcb2,
                                              float* __restrict__ out, int N) {
  __shared__ float g[256];
  __shared__ float hid[64];
  int t = threadIdx.x;
  g[t] = pool[t] * (1.0f / (float)N);
  __syncthreads();
  if (t < 64) {
    float s = fcb1[t];
    for (int f = 0; f < 256; ++f) s += g[f] * fcW1[f * 64 + t];
    hid[t] = leaky(s);
  }
  __syncthreads();
  if (t == 0) {
    float s = fcb2[0];
    for (int j = 0; j < 64; ++j) s += hid[j] * fcW2[j];
    out[0] = 1.0f / (1.0f + __expf(-s));
  }
}

// ---------- launch ----------
extern "C" void kernel_launch(void* const* d_in, const int* in_sizes, int n_in,
                              void* d_out, int out_size, void* d_ws, size_t ws_size,
                              hipStream_t stream) {
  (void)n_in; (void)out_size; (void)ws_size;
  const float* x    = (const float*)d_in[0];
  const int*   ei   = (const int*)d_in[1];   // JAX x64 disabled -> int32 [2][E]
  const float* Ws[4] = {(const float*)d_in[2], (const float*)d_in[4],
                        (const float*)d_in[6], (const float*)d_in[8]};
  const float* bs[4] = {(const float*)d_in[3], (const float*)d_in[5],
                        (const float*)d_in[7], (const float*)d_in[9]};
  const float* fcW1 = (const float*)d_in[10];
  const float* fcb1 = (const float*)d_in[11];
  const float* fcW2 = (const float*)d_in[12];
  const float* fcb2 = (const float*)d_in[13];

  const int N = in_sizes[0] / 512;
  const long long E = in_sizes[1] / 2;

  size_t off = 0;
  auto carve = [&](size_t bytes) -> void* {
    void* p = (char*)d_ws + off;
    off += (bytes + 255) & ~(size_t)255;
    return p;
  };
  unsigned short* Xb0 = (unsigned short*)carve((size_t)N * 512 * 2);
  unsigned short* Xb1 = (unsigned short*)carve((size_t)N * 512 * 2);
  float* xw   = (float*)carve((size_t)N * 512 * 4);
  float* agg  = (float*)carve((size_t)N * 512 * 4);
  unsigned short* Wt = (unsigned short*)carve((size_t)512 * 512 * 2);
  float* deg  = (float*)carve((size_t)N * 4);
  float* dis  = (float*)carve((size_t)N * 4);
  float* pool = (float*)carve(256 * 4);

  auto nb = [](long long n, int b) { return (unsigned)((n + b - 1) / b); };

  k_zero_f32<<<nb(N, 256), 256, 0, stream>>>(deg, N);
  k_zero_f32<<<1, 256, 0, stream>>>(pool, 256);
  k_degree<<<nb(E, 256), 256, 0, stream>>>(ei, deg, E);
  k_dis<<<nb(N, 256), 256, 0, stream>>>(deg, dis, N);
  k_f32_to_bf16<<<nb((long long)N * 512, 256), 256, 0, stream>>>(x, Xb0, (long long)N * 512);

  struct Layer { const float* W; const float* b; int fin; int fout;
                 unsigned short* in; unsigned short* out; };
  Layer L[4] = {
    {Ws[0], bs[0], 512, 512, Xb0, Xb1},
    {Ws[1], bs[1], 512, 512, Xb1, Xb0},
    {Ws[2], bs[2], 512, 512, Xb0, Xb1},
    {Ws[3], bs[3], 512, 256, Xb1, Xb0},
  };

  for (int l = 0; l < 4; ++l) {
    const int fin = L[l].fin, fout = L[l].fout;
    const int fs = (fout == 512) ? 9 : 8;
    k_wtrans<<<nb((long long)fin * fout, 256), 256, 0, stream>>>(L[l].W, Wt, fin, fout);
    dim3 g((N + 63) / 64, fout / 64);
    k_gemm_bf16<<<g, 256, 0, stream>>>(L[l].in, Wt, xw, N, fin, fout);
    const long long tot = (long long)N << fs;
    k_init_agg<<<nb(tot, 256), 256, 0, stream>>>(xw, dis, L[l].b, agg, tot, fs);
    k_scatter<<<nb(E << (fs - 2), 256), 256, 0, stream>>>(ei, xw, dis, agg, E, fs - 2);
    k_finalize<<<nb(tot, 256), 256, 0, stream>>>(agg, L[l].out, tot);
  }

  k_pool<<<nb(N, 256), 256, 0, stream>>>(Xb0, pool, N);
  k_head<<<1, 256, 0, stream>>>(pool, fcW1, fcb1, fcW2, fcb2, (float*)d_out, N);
}
